// LSTM_20194936226426
// MI455X (gfx1250) — compile-verified, matrix operand order checked by reading, not run on hardware
//
#include <hip/hip_runtime.h>

// ---------------------------------------------------------------------------
// LSTM on MI455X (gfx1250): bf16 WMMA GEMMs, persistent recurrent kernel with
// LDS-resident W_hh, register-resident cell state, fully-unrolled K loops
// (immediate-offset addressing -> no WMMA/VALU WAR hazard NOPs).
// ---------------------------------------------------------------------------

typedef __attribute__((ext_vector_type(16))) __bf16 v16bf;
typedef __attribute__((ext_vector_type(8)))  __bf16 v8bf;
typedef __attribute__((ext_vector_type(8)))  float  v8f;

#define T_STEPS 1024
#define BATCH   64
#define IDIM    512
#define HDIM    512
#define GDIM    (4 * HDIM)   // 2048 gate columns
#define NWG     16           // workgroups in recurrent kernel
#define HS      (HDIM / NWG) // 32 h-columns per WG
#define LDS_STRIDE 520       // 512 + 8 bf16 pad -> 1040B, 16B aligned, conflict-free

// Build a 16x32 bf16 WMMA A/B fragment from a K-contiguous row/column.
// Lanes 0-15 hold K = {k0..k0+7, k0+16..k0+23}; lanes 16-31 the +8 half
// (caller folds hlf*8 into p).
__device__ __forceinline__ v16bf load_frag(const __bf16* p) {
    v8bf lo = *(const v8bf*)(p);
    v8bf hi = *(const v8bf*)(p + 16);
    v16bf r;
#pragma unroll
    for (int i = 0; i < 8; ++i) { r[i] = lo[i]; r[i + 8] = hi[i]; }
    return r;
}

__device__ __forceinline__ float sigmoidf_fast(float x) {
    return 1.0f / (1.0f + __expf(-x));
}

// ---------------------------------------------------------------------------
// fp32 -> bf16 conversion (grid-stride)
// ---------------------------------------------------------------------------
__global__ void cvt_f32_bf16(const float* __restrict__ src,
                             __bf16* __restrict__ dst, int n) {
    for (int i = blockIdx.x * blockDim.x + threadIdx.x; i < n;
         i += gridDim.x * blockDim.x)
        dst[i] = (__bf16)src[i];
}

// ---------------------------------------------------------------------------
// Phase 1: x_gates[T*B, 2048] = X[T*B,512] @ Wih^T + (bias_ih + bias_hh)
// Block: 256 threads = 8 waves; block tile 128(M) x 128(N).
// Wave (mt = w>>1, ch = w&1) computes 32 rows x 64 cols = 2x4 WMMA tiles.
// K loop fully unrolled: all loads use immediate offsets from fixed bases.
// ---------------------------------------------------------------------------
__global__ __launch_bounds__(256)
void gemm_xproj(const __bf16* __restrict__ X,    // [65536, 512]
                const __bf16* __restrict__ Wih,  // [2048, 512] (rows = gate cols)
                const float*  __restrict__ bias_ih,
                const float*  __restrict__ bias_hh,
                float* __restrict__ xg) {        // [65536, 2048]
    const int lane = threadIdx.x & 31;
    const int wave = threadIdx.x >> 5;
    const int hlf  = lane >> 4;
    const int l16  = lane & 15;
    const int mt   = wave >> 1;   // 0..3 -> 32-row tile
    const int ch   = wave & 1;    // 0..1 -> 64-col half

    const int mbase = blockIdx.x * 128 + mt * 32;   // 2 sub-tiles of 16 rows
    const int nbase = blockIdx.y * 128 + ch * 64;   // 4 col tiles of 16

    const __bf16* Arow0 = X + (size_t)(mbase + l16) * IDIM + hlf * 8;
    const __bf16* Arow1 = Arow0 + 16 * IDIM;
    const __bf16* Bcol[4];
#pragma unroll
    for (int j = 0; j < 4; ++j)
        Bcol[j] = Wih + (size_t)(nbase + j * 16 + l16) * IDIM + hlf * 8;

    v8f acc[2][4] = {};
#pragma unroll
    for (int kk = 0; kk < IDIM; kk += 32) {
        v16bf a0 = load_frag(Arow0 + kk);
        v16bf a1 = load_frag(Arow1 + kk);
        v16bf b0 = load_frag(Bcol[0] + kk);
        v16bf b1 = load_frag(Bcol[1] + kk);
        v16bf b2 = load_frag(Bcol[2] + kk);
        v16bf b3 = load_frag(Bcol[3] + kk);
        acc[0][0] = __builtin_amdgcn_wmma_f32_16x16x32_bf16(
            false, a0, false, b0, (short)0, acc[0][0], false, false);
        acc[0][1] = __builtin_amdgcn_wmma_f32_16x16x32_bf16(
            false, a0, false, b1, (short)0, acc[0][1], false, false);
        acc[0][2] = __builtin_amdgcn_wmma_f32_16x16x32_bf16(
            false, a0, false, b2, (short)0, acc[0][2], false, false);
        acc[0][3] = __builtin_amdgcn_wmma_f32_16x16x32_bf16(
            false, a0, false, b3, (short)0, acc[0][3], false, false);
        acc[1][0] = __builtin_amdgcn_wmma_f32_16x16x32_bf16(
            false, a1, false, b0, (short)0, acc[1][0], false, false);
        acc[1][1] = __builtin_amdgcn_wmma_f32_16x16x32_bf16(
            false, a1, false, b1, (short)0, acc[1][1], false, false);
        acc[1][2] = __builtin_amdgcn_wmma_f32_16x16x32_bf16(
            false, a1, false, b2, (short)0, acc[1][2], false, false);
        acc[1][3] = __builtin_amdgcn_wmma_f32_16x16x32_bf16(
            false, a1, false, b3, (short)0, acc[1][3], false, false);
    }

#pragma unroll
    for (int j = 0; j < 4; ++j) {
        const int ncol = nbase + j * 16 + l16;
        const float bsum = bias_ih[ncol] + bias_hh[ncol];
#pragma unroll
        for (int ms = 0; ms < 2; ++ms)
#pragma unroll
            for (int r = 0; r < 8; ++r) {
                const int m = mbase + ms * 16 + r + hlf * 8;
                xg[(size_t)m * GDIM + ncol] = acc[ms][j][r] + bsum;
            }
    }
}

// ---------------------------------------------------------------------------
// Phase 2: persistent recurrent scan. NWG=16 WGs x 256 threads.
// WG `wg` owns h columns [wg*32, wg*32+32) and the 4 matching gate blocks.
// W_hh slice (4 gates x 32 cols x 512 K, bf16) lives in LDS for all 1024 steps.
// Wave (mt = w>>1, ch = w&1) owns rows mt*16..+15, cols hbase+ch*16..+15 and
// keeps c for that tile in 8 VGPRs across the whole scan.
// ---------------------------------------------------------------------------
__global__ __launch_bounds__(256)
void lstm_scan(const float* __restrict__ Whh,   // [2048, 512] fp32
               const float* __restrict__ xg,    // [65536, 2048] fp32 (w/ biases)
               const float* __restrict__ init_c,// [64, 512]
               __bf16* __restrict__ hbuf,       // [2][64][512] bf16 ping-pong
               float* __restrict__ h_out,       // [64, 512] final output
               unsigned* __restrict__ bar) {
    extern __shared__ __bf16 ldsW[];            // 128 cols * 520 = 130 KB

    const int lane = threadIdx.x & 31;
    const int wave = threadIdx.x >> 5;
    const int hlf  = lane >> 4;
    const int l16  = lane & 15;
    const int mt   = wave >> 1;      // 0..3
    const int ch   = wave & 1;       // 0..1
    const int wg   = blockIdx.x;     // 0..15
    const int hbase = wg * HS;

    // Stage W_hh slice into LDS as bf16 (K contiguous per column, padded stride).
    // colIdx = g*32 + c  <->  global row g*512 + hbase + c.
    for (int e = threadIdx.x; e < 128 * 512; e += 256) {
        const int colIdx = e >> 9;
        const int k      = e & 511;
        const int grow   = (colIdx >> 5) * HDIM + hbase + (colIdx & 31);
        ldsW[colIdx * LDS_STRIDE + k] = (__bf16)Whh[(size_t)grow * HDIM + k];
    }
    __syncthreads();

    // Cell state for this wave's 16x16 (b, h) tile: 8 fp32 per lane.
    const int n_h = hbase + ch * 16 + l16;
    v8f c = {};
#pragma unroll
    for (int r = 0; r < 8; ++r) {
        const int b = mt * 16 + r + hlf * 8;
        c[r] = init_c[b * HDIM + n_h];
    }

    const __bf16* ldsB0 = &ldsW[(0 * 32 + ch * 16 + l16) * LDS_STRIDE + hlf * 8];
    const __bf16* ldsB1 = &ldsW[(1 * 32 + ch * 16 + l16) * LDS_STRIDE + hlf * 8];
    const __bf16* ldsB2 = &ldsW[(2 * 32 + ch * 16 + l16) * LDS_STRIDE + hlf * 8];
    const __bf16* ldsB3 = &ldsW[(3 * 32 + ch * 16 + l16) * LDS_STRIDE + hlf * 8];

    for (int t = 0; t < T_STEPS; ++t) {
        const __bf16* hin  = hbuf + (size_t)(t & 1) * (BATCH * HDIM);
        __bf16*       hout = hbuf + (size_t)((t + 1) & 1) * (BATCH * HDIM);

        // Preload this step's x_gates values so they complete under the WMMAs,
        // and prefetch next step's slice into cache (global_prefetch_b8).
        const float* xgt = xg + (size_t)t * BATCH * GDIM;
        float xgv[4][8];
#pragma unroll
        for (int r = 0; r < 8; ++r) {
            const int b = mt * 16 + r + hlf * 8;
            const float* xr = xgt + (size_t)b * GDIM + n_h;
#pragma unroll
            for (int g = 0; g < 4; ++g) xgv[g][r] = xr[g * HDIM];
        }
        if (t + 1 < T_STEPS) {
            const float* xgn = xgt + (size_t)BATCH * GDIM;
#pragma unroll
            for (int r = 0; r < 8; ++r) {
                const int b = mt * 16 + r + hlf * 8;
                const float* xr = xgn + (size_t)b * GDIM + n_h;
#pragma unroll
                for (int g = 0; g < 4; ++g)
                    __builtin_prefetch(xr + g * HDIM, 0, 3);
            }
        }

        // gates slice: acc[g] = h @ Whh_slice(g)^T. Fully unrolled K loop:
        // A from L2-resident h buffer, B from LDS, immediate offsets only.
        const __bf16* Ah = hin + (size_t)(mt * 16 + l16) * HDIM + hlf * 8;
        v8f acc[4] = {};
#pragma unroll
        for (int kk = 0; kk < HDIM; kk += 32) {
            v16bf a  = load_frag(Ah + kk);
            v16bf b0 = load_frag(ldsB0 + kk);
            v16bf b1 = load_frag(ldsB1 + kk);
            v16bf b2 = load_frag(ldsB2 + kk);
            v16bf b3 = load_frag(ldsB3 + kk);
            acc[0] = __builtin_amdgcn_wmma_f32_16x16x32_bf16(
                false, a, false, b0, (short)0, acc[0], false, false);
            acc[1] = __builtin_amdgcn_wmma_f32_16x16x32_bf16(
                false, a, false, b1, (short)0, acc[1], false, false);
            acc[2] = __builtin_amdgcn_wmma_f32_16x16x32_bf16(
                false, a, false, b2, (short)0, acc[2], false, false);
            acc[3] = __builtin_amdgcn_wmma_f32_16x16x32_bf16(
                false, a, false, b3, (short)0, acc[3], false, false);
        }

        // Elementwise LSTM update — fully register-local per wave.
#pragma unroll
        for (int r = 0; r < 8; ++r) {
            const int b = mt * 16 + r + hlf * 8;
            float ig = acc[0][r] + xgv[0][r];
            float fg = acc[1][r] + xgv[1][r];
            float cg = acc[2][r] + xgv[2][r];
            float og = acc[3][r] + xgv[3][r];
            ig = sigmoidf_fast(ig);
            fg = sigmoidf_fast(fg);
            og = sigmoidf_fast(og);
            cg = tanhf(cg);
            const float cy = fg * c[r] + ig * cg;
            c[r] = cy;
            const float hy = og * tanhf(cy);
            hout[b * HDIM + n_h] = (__bf16)hy;
            if (t == T_STEPS - 1) h_out[b * HDIM + n_h] = hy;
        }

        // Device-wide barrier (16 WGs): release h writes, then rendezvous.
        __threadfence();
        __syncthreads();
        if (threadIdx.x == 0) {
            atomicAdd(bar, 1u);
            const unsigned target = (unsigned)(t + 1) * NWG;
            while (__hip_atomic_load(bar, __ATOMIC_ACQUIRE,
                                     __HIP_MEMORY_SCOPE_AGENT) < target) {
                __builtin_amdgcn_s_sleep(1);
            }
        }
        __syncthreads();
    }
}

// ---------------------------------------------------------------------------
// Host-side launcher. Workspace layout (bytes):
//   [0)      xg         : 65536*2048*4 = 536,870,912
//   [+ )     input bf16 : 33,554,432*2 =  67,108,864
//   [+ )     Wih bf16   :  1,048,576*2 =   2,097,152
//   [+ )     hbuf bf16  : 2*64*512*2   =     131,072
//   [+ )     barrier    : 4
// ---------------------------------------------------------------------------
extern "C" void kernel_launch(void* const* d_in, const int* in_sizes, int n_in,
                              void* d_out, int out_size, void* d_ws, size_t ws_size,
                              hipStream_t stream) {
    const float* input     = (const float*)d_in[0]; // [1024, 64, 512]
    const float* initial_h = (const float*)d_in[1]; // [64, 512]
    const float* initial_c = (const float*)d_in[2]; // [64, 512]
    const float* weight_ih = (const float*)d_in[3]; // [2048, 512]
    const float* weight_hh = (const float*)d_in[4]; // [2048, 512]
    const float* bias_ih   = (const float*)d_in[5]; // [2048]
    const float* bias_hh   = (const float*)d_in[6]; // [2048]
    float*       out       = (float*)d_out;         // [64, 512]

    char* ws = (char*)d_ws;
    const size_t xg_bytes  = (size_t)T_STEPS * BATCH * GDIM * sizeof(float);
    const size_t xin_elems = (size_t)T_STEPS * BATCH * IDIM;
    const size_t wih_elems = (size_t)GDIM * IDIM;

    float*    xg       = (float*)ws;
    __bf16*   x_bf16   = (__bf16*)(ws + xg_bytes);
    __bf16*   wih_bf16 = (__bf16*)(ws + xg_bytes + xin_elems * 2);
    __bf16*   hbuf     = (__bf16*)(ws + xg_bytes + xin_elems * 2 + wih_elems * 2);
    unsigned* bar      = (unsigned*)(ws + xg_bytes + xin_elems * 2 + wih_elems * 2
                                     + 2u * BATCH * HDIM * 2);

    // Reset the grid barrier counter (graph-capture-safe async memset).
    hipMemsetAsync(bar, 0, sizeof(unsigned), stream);

    // fp32 -> bf16 conversions.
    cvt_f32_bf16<<<4096, 256, 0, stream>>>(input, x_bf16, (int)xin_elems);
    cvt_f32_bf16<<<1024, 256, 0, stream>>>(weight_ih, wih_bf16, (int)wih_elems);
    cvt_f32_bf16<<<128, 256, 0, stream>>>(initial_h, hbuf, BATCH * HDIM);

    // Phase 1: big input-projection GEMM (+ fused biases).
    dim3 g1((T_STEPS * BATCH) / 128, GDIM / 128);
    gemm_xproj<<<g1, 256, 0, stream>>>(x_bf16, wih_bf16, bias_ih, bias_hh, xg);

    // Phase 2: persistent recurrent scan (dynamic LDS: 128 cols * 520 * 2B).
    const size_t lds_bytes = (size_t)128 * LDS_STRIDE * sizeof(__bf16);
    lstm_scan<<<NWG, 256, lds_bytes, stream>>>(weight_hh, xg, initial_c,
                                               hbuf, out, bar);
}
